// OuterProductMean_52183852646492
// MI455X (gfx1250) — matrix operand order repaired and checked by reading, not compiled, not run on hardware
//
#include <hip/hip_runtime.h>
#include <hip/hip_bf16.h>

// Sizes fixed by the reference.
#define S_DIM 256
#define N_DIM 512
#define C_DIM 32

typedef _Float16 h8   __attribute__((ext_vector_type(8)));
typedef _Float16 v16h __attribute__((ext_vector_type(16)));
typedef float    v8f  __attribute__((ext_vector_type(8)));

// ---------------------------------------------------------------------------
// CDNA5 async copy: global memory -> LDS, one b128 per lane, ASYNCcnt-tracked.
// lds_addr = LDS byte offset (low 32 bits of a generic pointer into __shared__).
// ---------------------------------------------------------------------------
__device__ __forceinline__ void async_copy_b128(unsigned int lds_addr,
                                                const void* gaddr) {
    asm volatile("global_load_async_to_lds_b128 %0, %1, off"
                 :: "v"(lds_addr), "v"(gaddr)
                 : "memory");
}

__device__ __forceinline__ void wait_asynccnt0() {
    asm volatile("s_wait_asynccnt 0x0" ::: "memory");
}

// ---------------------------------------------------------------------------
// WMMA fragment loaders (layouts per CDNA5 ISA 7.12.2, f16 16x16x32)
// A operand: matrix stored row-major [M][K]; lane L: M=L%16,
//   halves K = kb..kb+7 and 16+kb..16+kb+7 with kb = (L<16?0:8).
// B operand: B^T stored row-major [N][K]; lane L: N=L%16,
//   halves K = kb..kb+7 and 16+kb..16+kb+7 with kb = (L<16?0:8).
// ---------------------------------------------------------------------------
__device__ __forceinline__ v16h load_frag_A(const _Float16* base, int row,
                                            int rowStride, int k0, int lane) {
    int kb = (lane < 16) ? 0 : 8;
    const _Float16* p = base + row * rowStride + k0 + kb;
    h8 lo = *(const h8*)(p);
    h8 hi = *(const h8*)(p + 16);
    return __builtin_shufflevector(lo, hi, 0,1,2,3,4,5,6,7,8,9,10,11,12,13,14,15);
}

__device__ __forceinline__ v16h load_frag_B(const _Float16* base, int row,
                                            int rowStride, int k0, int lane) {
    int kb = (lane < 16) ? 0 : 8;
    const _Float16* p = base + row * rowStride + k0 + kb;
    h8 lo = *(const h8*)(p);
    h8 hi = *(const h8*)(p + 16);
    return __builtin_shufflevector(lo, hi, 0,1,2,3,4,5,6,7,8,9,10,11,12,13,14,15);
}

// ---------------------------------------------------------------------------
// Stage 1: fused LayerNorm + dual projection + mask, transposed f16 outputs
//   aT[n][c][s] = ((LN(m[s,n,:]) @ Wa + ba)[c]) * mask[s,n]   (f16)
//   bT[n][e][s] likewise with Wb/bb.
// grid = (N_DIM), block = S_DIM threads (one thread per sequence s).
// ---------------------------------------------------------------------------
__global__ void __launch_bounds__(256) opm_ln_proj(
    const float* __restrict__ m, const float* __restrict__ mask,
    const float* __restrict__ gamma, const float* __restrict__ beta,
    const float* __restrict__ Wa, const float* __restrict__ ba,
    const float* __restrict__ Wb, const float* __restrict__ bb,
    _Float16* __restrict__ aT, _Float16* __restrict__ bT)
{
    __shared__ float sWa[1024], sWb[1024], sba[32], sbb[32], sg[32], sbe[32];
    const int t = threadIdx.x;
    for (int i = t; i < 1024; i += 256) { sWa[i] = Wa[i]; sWb[i] = Wb[i]; }
    if (t < 32) { sba[t] = ba[t]; sbb[t] = bb[t]; sg[t] = gamma[t]; sbe[t] = beta[t]; }
    __syncthreads();

    const int n = blockIdx.x;   // residue
    const int s = t;            // sequence
    const float* row = m + ((size_t)s * N_DIM + n) * C_DIM;

    float x[32];
#pragma unroll
    for (int c = 0; c < 8; ++c) {
        float4 v = ((const float4*)row)[c];
        x[4*c+0] = v.x; x[4*c+1] = v.y; x[4*c+2] = v.z; x[4*c+3] = v.w;
    }
    float mu = 0.f;
#pragma unroll
    for (int c = 0; c < 32; ++c) mu += x[c];
    mu *= (1.f / 32.f);
    float var = 0.f;
#pragma unroll
    for (int c = 0; c < 32; ++c) { float d = x[c] - mu; var += d * d; }
    var *= (1.f / 32.f);
    const float rs = rsqrtf(var + 1e-5f);

    float mh[32];
#pragma unroll
    for (int c = 0; c < 32; ++c) mh[c] = (x[c] - mu) * rs * sg[c] + sbe[c];

    const float mval = mask[(size_t)s * N_DIM + n];
    const size_t obase = ((size_t)n * 32) * 256 + s;

    for (int c = 0; c < 32; ++c) {
        float sa = sba[c], sb = sbb[c];
#pragma unroll
        for (int k = 0; k < 32; ++k) {
            sa += mh[k] * sWa[k * 32 + c];
            sb += mh[k] * sWb[k * 32 + c];
        }
        aT[obase + (size_t)c * 256] = (_Float16)(sa * mval);
        bT[obase + (size_t)c * 256] = (_Float16)(sb * mval);
    }
}

// ---------------------------------------------------------------------------
// Stage 1b: WoutT[k][q] = Wout[q][k] in f16 (B-operand layout for projection)
// ---------------------------------------------------------------------------
__global__ void __launch_bounds__(256) opm_wtrans(
    const float* __restrict__ Wout, _Float16* __restrict__ WoutT)
{
    int idx = blockIdx.x * 256 + threadIdx.x;   // 32*1024 total
    int k = idx >> 10, q = idx & 1023;
    WoutT[idx] = (_Float16)Wout[q * 32 + k];
}

// ---------------------------------------------------------------------------
// Stage 1c: normInv[i][j] = 1 / (sum_s mask[s,i]*mask[s,j] + 1e-3)
// grid = (N_DIM, 2), block = 256
// ---------------------------------------------------------------------------
__global__ void __launch_bounds__(256) opm_norm(
    const float* __restrict__ mask, float* __restrict__ normInv)
{
    __shared__ float mi[256];
    const int i = blockIdx.x;
    const int j = blockIdx.y * 256 + threadIdx.x;
    mi[threadIdx.x] = mask[(size_t)threadIdx.x * N_DIM + i];
    __syncthreads();
    float acc = 0.f;
    for (int s = 0; s < 256; ++s) acc += mi[s] * mask[(size_t)s * N_DIM + j];
    normInv[(size_t)i * N_DIM + j] = 1.0f / (acc + 1e-3f);
}

// ---------------------------------------------------------------------------
// Stage 2: fused outer-product GEMM + Wout projection + norm epilogue.
// Block = 256 threads (8 waves), owns a 4i x 4j tile of residue pairs.
// LDS: sA 64KB | sB 64KB | sO 32KB | sAcc 2KB  (162KB of the 320KB WGP LDS)
// Tile staging uses CDNA5 async global->LDS copies (ASYNCcnt).
// ---------------------------------------------------------------------------
__global__ void __launch_bounds__(256) opm_main(
    const _Float16* __restrict__ aT, const _Float16* __restrict__ bT,
    const _Float16* __restrict__ WoutT, const float* __restrict__ normInv,
    const float* __restrict__ bout, float* __restrict__ out)
{
    extern __shared__ char smem[];
    _Float16* sA   = (_Float16*)(smem);                 // 4 tiles * 32 * 256 f16
    _Float16* sB   = (_Float16*)(smem + 65536);         // 4 tiles * 32 * 256 f16
    _Float16* sO   = (_Float16*)(smem + 131072);        // 16 pairs * 1024 f16
    float*    sAcc = (float*)   (smem + 163840);        // 16 pairs * 32 f32

    const int t    = threadIdx.x;
    const int lane = t & 31;
    const int w    = t >> 5;
    const int jBase = blockIdx.x * 4;
    const int iBase = blockIdx.y * 4;

    // ---- async-stage tiles into LDS (contiguous 64KB each), zero sAcc ----
    {
        const uint4* gA = (const uint4*)(aT + (size_t)iBase * 32 * 256) + t;
        const uint4* gB = (const uint4*)(bT + (size_t)jBase * 32 * 256) + t;
        // LDS byte offsets for this thread's b128 slots (low 32 bits of the
        // generic pointer are the LDS aperture offset).
        const unsigned int ldsA = (unsigned int)(uintptr_t)((uint4*)sA + t);
        const unsigned int ldsB = (unsigned int)(uintptr_t)((uint4*)sB + t);
#pragma unroll
        for (int r = 0; r < 16; ++r) {
            async_copy_b128(ldsA + r * 4096u, gA + r * 256);
            async_copy_b128(ldsB + r * 4096u, gB + r * 256);
        }
        sAcc[t * 2 + 0] = 0.f;
        sAcc[t * 2 + 1] = 0.f;
        wait_asynccnt0();       // data in LDS before signaling the barrier
    }
    __syncthreads();

    // ---- main contraction: O[c,e] = sum_s aT[i][c][s] * bT[j][e][s] ----
    const int iL = w & 3;          // local i
    const int j0 = (w >> 2) * 2;   // local j pair {j0, j0+1}

    v8f acc[2][2][2] = {};         // [jj][ct][et], each 16x16 f32 tile
    const _Float16* Abase = sA + (size_t)iL * 32 * 256;

#pragma unroll
    for (int ks = 0; ks < 8; ++ks) {
        const int k0 = ks * 32;
        v16h af[2];
#pragma unroll
        for (int ct = 0; ct < 2; ++ct)
            af[ct] = load_frag_A(Abase, ct * 16 + (lane & 15), 256, k0, lane);
#pragma unroll
        for (int jj = 0; jj < 2; ++jj) {
            const _Float16* Bbase = sB + (size_t)(j0 + jj) * 32 * 256;
#pragma unroll
            for (int et = 0; et < 2; ++et) {
                v16h bf = load_frag_B(Bbase, et * 16 + (lane & 15), 256, k0, lane);
#pragma unroll
                for (int ct = 0; ct < 2; ++ct)
                    acc[jj][ct][et] = __builtin_amdgcn_wmma_f32_16x16x32_f16(
                        false, af[ct], false, bf, (short)0, acc[jj][ct][et],
                        false, false);
            }
        }
    }

    // ---- spill outer tiles to LDS as f16 rows of the projection A matrix ----
#pragma unroll
    for (int jj = 0; jj < 2; ++jj) {
        const int p = iL * 4 + (j0 + jj);       // pair row 0..15
        _Float16* op = sO + (size_t)p * 1024;
#pragma unroll
        for (int ct = 0; ct < 2; ++ct)
#pragma unroll
            for (int et = 0; et < 2; ++et)
#pragma unroll
                for (int v = 0; v < 8; ++v) {
                    const int c = ct * 16 + v + ((lane >= 16) ? 8 : 0);
                    const int e = et * 16 + (lane & 15);
                    op[c * 32 + e] = (_Float16)acc[jj][ct][et][v];
                }
    }
    __syncthreads();

    // ---- projection: out[p,k] = sum_q sO[p][q] * WoutT[k][q] ----
    // wave w covers q in [w*128, w*128+128); partials reduced via ds_add_f32
    {
        v8f pacc[2] = {};
#pragma unroll
        for (int kk = 0; kk < 4; ++kk) {
            const int q0 = w * 128 + kk * 32;
            v16h of = load_frag_A(sO, (lane & 15), 1024, q0, lane);
#pragma unroll
            for (int kt = 0; kt < 2; ++kt) {
                v16h wf = load_frag_B(WoutT, kt * 16 + (lane & 15), 1024, q0, lane);
                pacc[kt] = __builtin_amdgcn_wmma_f32_16x16x32_f16(
                    false, of, false, wf, (short)0, pacc[kt], false, false);
            }
        }
#pragma unroll
        for (int kt = 0; kt < 2; ++kt)
#pragma unroll
            for (int v = 0; v < 8; ++v) {
                const int p = v + ((lane >= 16) ? 8 : 0);
                const int k = kt * 16 + (lane & 15);
                atomicAdd(&sAcc[p * 32 + k], pacc[kt][v]);
            }
    }
    __syncthreads();

    // ---- epilogue: (proj + bout) * normInv, write out[i][j][k] ----
    {
        int idx = t * 2;
#pragma unroll
        for (int u = 0; u < 2; ++u, ++idx) {
            const int p = idx >> 5, k = idx & 31;
            const int i = iBase + (p >> 2);
            const int j = jBase + (p & 3);
            const float val = (sAcc[idx] + bout[k]) * normInv[(size_t)i * N_DIM + j];
            out[((size_t)i * N_DIM + j) * C_DIM + k] = val;
        }
    }
}

// ---------------------------------------------------------------------------
// Launch
// ---------------------------------------------------------------------------
extern "C" void kernel_launch(void* const* d_in, const int* in_sizes, int n_in,
                              void* d_out, int out_size, void* d_ws, size_t ws_size,
                              hipStream_t stream) {
    const float* m     = (const float*)d_in[0];
    const float* mask  = (const float*)d_in[1];
    const float* g     = (const float*)d_in[2];
    const float* be    = (const float*)d_in[3];
    const float* Wa    = (const float*)d_in[4];
    const float* ba    = (const float*)d_in[5];
    const float* Wb    = (const float*)d_in[6];
    const float* bb    = (const float*)d_in[7];
    const float* Wout  = (const float*)d_in[8];
    const float* bout  = (const float*)d_in[9];
    float* out = (float*)d_out;

    char* ws = (char*)d_ws;
    _Float16* aT      = (_Float16*)(ws);              // 512*32*256 f16 = 8 MiB
    _Float16* bT      = (_Float16*)(ws + 8388608);    // 8 MiB
    _Float16* WoutT   = (_Float16*)(ws + 16777216);   // 64 KiB
    float*    normInv = (float*)   (ws + 16842752);   // 1 MiB

    opm_ln_proj<<<dim3(N_DIM), 256, 0, stream>>>(m, mask, g, be, Wa, ba, Wb, bb, aT, bT);
    opm_wtrans <<<dim3(128),  256, 0, stream>>>(Wout, WoutT);
    opm_norm   <<<dim3(N_DIM, 2), 256, 0, stream>>>(mask, normInv);

    const size_t lds = 65536 + 65536 + 32768 + 2048;  // 162 KB of 320 KB WGP LDS
    opm_main<<<dim3(N_DIM / 4, N_DIM / 4), 256, lds, stream>>>(
        aT, bT, WoutT, normInv, bout, out);
}
